// EGNNet_40458591929210
// MI455X (gfx1250) — matrix operand order repaired and checked by reading, not compiled, not run on hardware
//
#include <hip/hip_runtime.h>
#include <hip/hip_bf16.h>

#define NNODES 100000
#define NEDGES 1600000
#define NGRAPHS 128

typedef __attribute__((ext_vector_type(16))) _Float16 v16h;
typedef __attribute__((ext_vector_type(8)))  _Float16 v8h;
typedef __attribute__((ext_vector_type(8)))  float    v8f;

__device__ __forceinline__ float silu_f(float x) {
    return x / (1.0f + __expf(-x));
}

// WMMA D = A(16x32 f16) * B(32x16 f16) + C(16x16 f32)
__device__ __forceinline__ v8f wmma_f16(v16h a, v16h b, v8f c) {
    return __builtin_amdgcn_wmma_f32_16x16x32_f16(false, a, false, b, (short)0, c, false, false);
}

// A fragment from row-major LDS buffer [rows][strideH] halves.
// ISA 16-bit A 16x32 layout: lane l in [0,16): row=l, K=k0..k0+7 and K=k0+16..k0+23,
// with k0 = kt*32 (+8 for lanes 16-31).
__device__ __forceinline__ v16h frag_a_lds(const _Float16* buf, int strideH, int kt, int lane) {
    int row = lane & 15;
    int k0  = kt * 32 + ((lane >> 4) << 3);
    const _Float16* p = buf + row * strideH + k0;
    v8h lo = *(const v8h*)(p);
    v8h hi = *(const v8h*)(p + 16);
    v16h a;
#pragma unroll
    for (int i = 0; i < 8; ++i) { a[i] = lo[i]; a[i + 8] = hi[i]; }
    return a;
}

// B fragment: pre-swizzled in global memory, 512 contiguous halves per fragment,
// lane-major (lane l holds its 16 halves contiguously) -> one 32B load per lane.
__device__ __forceinline__ v16h frag_b_glob(const _Float16* fragbase, int lane) {
    return ((const v16h*)fragbase)[lane];
}

// ---------------------------------------------------------------------------
// Conversion kernels (run once per launch, cheap)
// ---------------------------------------------------------------------------
__global__ void k_h_to_f16(const float* __restrict__ x, _Float16* __restrict__ h16, long n) {
    long i = (long)blockIdx.x * 256 + threadIdx.x;
    if (i < n) h16[i] = (_Float16)x[i];
}

__global__ void k_rel_dist(const float* __restrict__ pos, const int* __restrict__ srcI,
                           const int* __restrict__ dstI, float* __restrict__ rd) {
    long e = (long)blockIdx.x * 256 + threadIdx.x;
    if (e < NEDGES) {
        int s = srcI[e], d = dstI[e];
        float dx = pos[s * 3 + 0] - pos[d * 3 + 0];
        float dy = pos[s * 3 + 1] - pos[d * 3 + 1];
        float dz = pos[s * 3 + 2] - pos[d * 3 + 2];
        rd[e] = sqrtf(dx * dx + dy * dy + dz * dz);
    }
}

// Generic B-fragment conversion: src f32 row-major [K][N] -> f16 fragments [KT][NT][32][16]
// B layout (ISA 16-bit B 32x16): lane l: col = l&15, K rows kt*32 + (l>>4)*8 + (j&7) + (j>>3)*16, j=0..15.
__global__ void k_conv_bfrag(const float* __restrict__ src, _Float16* __restrict__ dst,
                             int K, int N, int KT, int NT) {
    long total = (long)KT * NT * 512;
    long i = (long)blockIdx.x * 256 + threadIdx.x;
    if (i >= total) return;
    int within = (int)(i & 511);
    int f = (int)(i >> 9);
    int lane = within >> 4, j = within & 15;
    int kt = f / NT, nt = f % NT;
    int k = kt * 32 + ((lane >> 4) << 3) + (j & 7) + ((j >> 3) << 4);
    int n = nt * 16 + (lane & 15);
    float v = (k < K && n < N) ? src[(long)k * N + n] : 0.0f;
    dst[i] = (_Float16)v;
}

// eW1 conversion: split into dst-part (rows 0..63) and src-part (rows 64..127);
// row 128 (rel_dist) is a rank-1 VALU epilogue. Per layer: [part][kt][nt][512],
// 258 cols zero-padded to 272 (17 n-tiles).
__global__ void k_conv_eW1(const float* __restrict__ eW1, _Float16* __restrict__ W1f) {
    long total = 3L * 2 * 2 * 17 * 512;
    long i = (long)blockIdx.x * 256 + threadIdx.x;
    if (i >= total) return;
    int within = (int)(i & 511);
    long f = i >> 9;
    int nt = (int)(f % 17); f /= 17;
    int kt = (int)(f % 2);  f /= 2;
    int part = (int)(f % 2); f /= 2;
    int layer = (int)f;
    int lane = within >> 4, j = within & 15;
    int k = kt * 32 + ((lane >> 4) << 3) + (j & 7) + ((j >> 3) << 4);   // 0..63
    int row = part * 64 + k;                    // 0..127
    int col = nt * 16 + (lane & 15);            // 0..271
    float v = (col < 258) ? eW1[((long)layer * 129 + row) * 258 + col] : 0.0f;
    W1f[i] = (_Float16)v;
}

// ---------------------------------------------------------------------------
// Edge MLP + scatter-add.  2 waves/block, each wave owns a 32-edge tile (M=32):
// each B fragment is loaded once and feeds TWO WMMAs -> halves L2 weight traffic.
// hid(32x272) = silu(hD@W1d + hS@W1s + d*w1row + b1); m = silu(hid@W2 + b2); atomic scatter.
// ---------------------------------------------------------------------------
__global__ __launch_bounds__(64) void k_edge(
    const _Float16* __restrict__ h16,
    const float* __restrict__ rel,
    const int* __restrict__ srcI, const int* __restrict__ dstI,
    const float* __restrict__ eW1l, const float* __restrict__ eb1l, const float* __restrict__ eb2l,
    const _Float16* __restrict__ W1f, const _Float16* __restrict__ W2f,
    float* __restrict__ m_acc)
{
    __shared__ __align__(16) _Float16 s_a[2][2][32][64];    // [wave][dst/src][row][feat]
    __shared__ __align__(16) _Float16 s_hid[2][32][288];    // padded hidden
    __shared__ float s_d[2][32];
    __shared__ int   s_idx[2][32];

    const int w = threadIdx.x >> 5;
    const int lane = threadIdx.x & 31;
    const long e0 = ((long)blockIdx.x * 2 + w) * 32;

    // stage gathered features: each row = 64 halves = 32 lanes x 4B
#pragma unroll 4
    for (int r = 0; r < 32; ++r) {
        long e = e0 + r;
        int dn = dstI[e];
        int sn = srcI[e];
        ((unsigned int*)&s_a[w][0][r][0])[lane] = ((const unsigned int*)(h16 + (long)dn * 64))[lane];
        ((unsigned int*)&s_a[w][1][r][0])[lane] = ((const unsigned int*)(h16 + (long)sn * 64))[lane];
    }
    {
        long e = e0 + lane;
        s_d[w][lane]   = rel[e];
        s_idx[w][lane] = dstI[e];
    }
    // zero pad region cols [272,288) for all 32 rows
    for (int i = lane; i < 512; i += 32)
        s_hid[w][i >> 4][272 + (i & 15)] = (_Float16)0.0f;

    const int colL  = lane & 15;
    const int rowHi = (lane >> 4) << 3;

    // GEMM1: 17 n-tiles, K = 64(dst) + 64(src), M = 32 (two row tiles per wave)
    for (int nt = 0; nt < 17; ++nt) {
        v8f acc0 = {}, acc1 = {};
#pragma unroll
        for (int part = 0; part < 2; ++part)
#pragma unroll
            for (int kt = 0; kt < 2; ++kt) {
                v16h b  = frag_b_glob(W1f + (((part * 2 + kt) * 17 + nt) << 9), lane);
                v16h a0 = frag_a_lds(&s_a[w][part][0][0],  64, kt, lane);
                v16h a1 = frag_a_lds(&s_a[w][part][16][0], 64, kt, lane);
                acc0 = wmma_f16(a0, b, acc0);
                acc1 = wmma_f16(a1, b, acc1);
            }
        int col = nt * 16 + colL;
        float w1r = (col < 258) ? eW1l[128 * 258 + col] : 0.0f;  // rel_dist row of eW1
        float bb  = (col < 258) ? eb1l[col] : 0.0f;
#pragma unroll
        for (int j = 0; j < 8; ++j) {
            int row0 = j + rowHi;
            int row1 = row0 + 16;
            float v0 = acc0[j] + s_d[w][row0] * w1r + bb;
            float v1 = acc1[j] + s_d[w][row1] * w1r + bb;
            s_hid[w][row0][col] = (_Float16)silu_f(v0);
            s_hid[w][row1][col] = (_Float16)silu_f(v1);
        }
    }

    // GEMM2: m(32x64) = silu(hid @ W2 + b2), K padded to 288; scatter-add to m_acc[dst]
    for (int nt = 0; nt < 4; ++nt) {
        v8f acc0 = {}, acc1 = {};
#pragma unroll
        for (int kt = 0; kt < 9; ++kt) {
            v16h b  = frag_b_glob(W2f + ((kt * 4 + nt) << 9), lane);
            v16h a0 = frag_a_lds(&s_hid[w][0][0],  288, kt, lane);
            v16h a1 = frag_a_lds(&s_hid[w][16][0], 288, kt, lane);
            acc0 = wmma_f16(a0, b, acc0);
            acc1 = wmma_f16(a1, b, acc1);
        }
        int col = nt * 16 + colL;
        float bb = eb2l[col];
#pragma unroll
        for (int j = 0; j < 8; ++j) {
            int row0 = j + rowHi;
            int row1 = row0 + 16;
            atomicAdd(&m_acc[(long)s_idx[w][row0] * 64 + col], silu_f(acc0[j] + bb));
            atomicAdd(&m_acc[(long)s_idx[w][row1] * 64 + col], silu_f(acc1[j] + bb));
        }
    }
}

// ---------------------------------------------------------------------------
// Node MLP: u=[h16, f16(m)] (32x128) -> silu(@nW1+b) (32x128) -> @nW2+b (32x64)
// 2 waves/block, 32 nodes per wave (M=32).
// ---------------------------------------------------------------------------
__global__ __launch_bounds__(64) void k_node(
    const _Float16* __restrict__ h16,
    const float* __restrict__ m_acc,
    const float* __restrict__ nb1l, const float* __restrict__ nb2l,
    const _Float16* __restrict__ nW1f, const _Float16* __restrict__ nW2f,
    _Float16* __restrict__ h16_out)
{
    __shared__ __align__(16) _Float16 s_u[2][32][128];
    __shared__ __align__(16) _Float16 s_h[2][32][128];

    const int w = threadIdx.x >> 5;
    const int lane = threadIdx.x & 31;
    const long n0 = ((long)blockIdx.x * 2 + w) * 32;

#pragma unroll 4
    for (int r = 0; r < 32; ++r) {
        long n = n0 + r;
        if (n < NNODES) {
            ((unsigned int*)&s_u[w][r][0])[lane] = ((const unsigned int*)(h16 + n * 64))[lane];
            float2 mv = ((const float2*)(m_acc + n * 64))[lane];
            s_u[w][r][64 + 2 * lane]     = (_Float16)mv.x;
            s_u[w][r][64 + 2 * lane + 1] = (_Float16)mv.y;
        } else {
            ((unsigned int*)&s_u[w][r][0])[lane] = 0u;
            s_u[w][r][64 + 2 * lane]     = (_Float16)0.0f;
            s_u[w][r][64 + 2 * lane + 1] = (_Float16)0.0f;
        }
    }

    const int colL  = lane & 15;
    const int rowHi = (lane >> 4) << 3;

    for (int nt = 0; nt < 8; ++nt) {
        v8f acc0 = {}, acc1 = {};
#pragma unroll
        for (int kt = 0; kt < 4; ++kt) {
            v16h b  = frag_b_glob(nW1f + ((kt * 8 + nt) << 9), lane);
            v16h a0 = frag_a_lds(&s_u[w][0][0],  128, kt, lane);
            v16h a1 = frag_a_lds(&s_u[w][16][0], 128, kt, lane);
            acc0 = wmma_f16(a0, b, acc0);
            acc1 = wmma_f16(a1, b, acc1);
        }
        int col = nt * 16 + colL;
        float bb = nb1l[col];
#pragma unroll
        for (int j = 0; j < 8; ++j) {
            s_h[w][j + rowHi][col]      = (_Float16)silu_f(acc0[j] + bb);
            s_h[w][j + rowHi + 16][col] = (_Float16)silu_f(acc1[j] + bb);
        }
    }

    for (int nt = 0; nt < 4; ++nt) {
        v8f acc0 = {}, acc1 = {};
#pragma unroll
        for (int kt = 0; kt < 4; ++kt) {
            v16h b  = frag_b_glob(nW2f + ((kt * 4 + nt) << 9), lane);
            v16h a0 = frag_a_lds(&s_h[w][0][0],  128, kt, lane);
            v16h a1 = frag_a_lds(&s_h[w][16][0], 128, kt, lane);
            acc0 = wmma_f16(a0, b, acc0);
            acc1 = wmma_f16(a1, b, acc1);
        }
        int col = nt * 16 + colL;
        float bb = nb2l[col];
#pragma unroll
        for (int j = 0; j < 8; ++j) {
            long na = n0 + j + rowHi;
            long nb = na + 16;
            if (na < NNODES) h16_out[na * 64 + col] = (_Float16)(acc0[j] + bb);
            if (nb < NNODES) h16_out[nb * 64 + col] = (_Float16)(acc1[j] + bb);
        }
    }
}

// ---------------------------------------------------------------------------
// Head per-node: t = silu(h@l1W+b1); o = t@l2W+b2; pool[batch[n]] += o   (M=32/wave)
// ---------------------------------------------------------------------------
__global__ __launch_bounds__(64) void k_head_node(
    const _Float16* __restrict__ h16,
    const int* __restrict__ batch,
    const float* __restrict__ l1b, const float* __restrict__ l2b,
    const _Float16* __restrict__ l1f, const _Float16* __restrict__ l2f,
    float* __restrict__ pool)
{
    __shared__ __align__(16) _Float16 s_a[2][32][64];
    __shared__ __align__(16) _Float16 s_t[2][32][64];
    __shared__ int s_b[2][32];

    const int w = threadIdx.x >> 5;
    const int lane = threadIdx.x & 31;
    const long n0 = ((long)blockIdx.x * 2 + w) * 32;

#pragma unroll 4
    for (int r = 0; r < 32; ++r) {
        long n = n0 + r;
        ((unsigned int*)&s_a[w][r][0])[lane] =
            (n < NNODES) ? ((const unsigned int*)(h16 + n * 64))[lane] : 0u;
    }
    {
        long n = n0 + lane;
        s_b[w][lane] = (n < NNODES) ? batch[n] : 0;
    }

    const int colL  = lane & 15;
    const int rowHi = (lane >> 4) << 3;

    for (int nt = 0; nt < 4; ++nt) {
        v8f acc0 = {}, acc1 = {};
#pragma unroll
        for (int kt = 0; kt < 2; ++kt) {
            v16h b  = frag_b_glob(l1f + ((kt * 4 + nt) << 9), lane);
            v16h a0 = frag_a_lds(&s_a[w][0][0],  64, kt, lane);
            v16h a1 = frag_a_lds(&s_a[w][16][0], 64, kt, lane);
            acc0 = wmma_f16(a0, b, acc0);
            acc1 = wmma_f16(a1, b, acc1);
        }
        int col = nt * 16 + colL;
        float bb = l1b[col];
#pragma unroll
        for (int j = 0; j < 8; ++j) {
            s_t[w][j + rowHi][col]      = (_Float16)silu_f(acc0[j] + bb);
            s_t[w][j + rowHi + 16][col] = (_Float16)silu_f(acc1[j] + bb);
        }
    }

    for (int nt = 0; nt < 4; ++nt) {
        v8f acc0 = {}, acc1 = {};
#pragma unroll
        for (int kt = 0; kt < 2; ++kt) {
            v16h b  = frag_b_glob(l2f + ((kt * 4 + nt) << 9), lane);
            v16h a0 = frag_a_lds(&s_t[w][0][0],  64, kt, lane);
            v16h a1 = frag_a_lds(&s_t[w][16][0], 64, kt, lane);
            acc0 = wmma_f16(a0, b, acc0);
            acc1 = wmma_f16(a1, b, acc1);
        }
        int col = nt * 16 + colL;
        float bb = l2b[col];
#pragma unroll
        for (int j = 0; j < 8; ++j) {
            long na = n0 + j + rowHi;
            long nb = na + 16;
            if (na < NNODES)
                atomicAdd(&pool[(long)s_b[w][j + rowHi] * 64 + col], acc0[j] + bb);
            if (nb < NNODES)
                atomicAdd(&pool[(long)s_b[w][j + rowHi + 16] * 64 + col], acc1[j] + bb);
        }
    }
}

// Tiny per-graph tail: out = silu(pool@l3W+b3)@l4W + b4   (128 graphs, fp32 scalar)
__global__ void k_head_graph(const float* __restrict__ pool,
                             const float* __restrict__ l3W, const float* __restrict__ l3b,
                             const float* __restrict__ l4W, const float* __restrict__ l4b,
                             float* __restrict__ out)
{
    int g = blockIdx.x * blockDim.x + threadIdx.x;
    if (g >= NGRAPHS) return;
    float o = l4b[0];
    for (int i = 0; i < 64; ++i) {
        float t = l3b[i];
        for (int k = 0; k < 64; ++k) t += pool[g * 64 + k] * l3W[k * 64 + i];
        o += silu_f(t) * l4W[i];
    }
    out[g] = o;
}

// ---------------------------------------------------------------------------
extern "C" void kernel_launch(void* const* d_in, const int* in_sizes, int n_in,
                              void* d_out, int out_size, void* d_ws, size_t ws_size,
                              hipStream_t stream)
{
    (void)in_sizes; (void)n_in; (void)out_size; (void)ws_size;
    const float* x    = (const float*)d_in[0];
    const float* pos  = (const float*)d_in[1];
    const int*   eidx = (const int*)d_in[2];
    const int*   batch= (const int*)d_in[3];
    const float* eW1  = (const float*)d_in[4];
    const float* eb1  = (const float*)d_in[5];
    const float* eW2  = (const float*)d_in[6];
    const float* eb2  = (const float*)d_in[7];
    const float* nW1  = (const float*)d_in[8];
    const float* nb1  = (const float*)d_in[9];
    const float* nW2  = (const float*)d_in[10];
    const float* nb2  = (const float*)d_in[11];
    const float* l1W  = (const float*)d_in[12];
    const float* l1b  = (const float*)d_in[13];
    const float* l2W  = (const float*)d_in[14];
    const float* l2b  = (const float*)d_in[15];
    const float* l3W  = (const float*)d_in[16];
    const float* l3b  = (const float*)d_in[17];
    const float* l4W  = (const float*)d_in[18];
    const float* l4b  = (const float*)d_in[19];
    float* out = (float*)d_out;

    const int* srcI = eidx;            // edge_index[0] = source
    const int* dstI = eidx + NEDGES;   // edge_index[1] = target

    // workspace carve-out (256B aligned)
    char* p = (char*)d_ws;
    auto take = [&](size_t bytes) -> char* {
        char* r = p; p += (bytes + 255) & ~(size_t)255; return r;
    };
    float*    rel   = (float*)take((size_t)NEDGES * 4);
    _Float16* h16a  = (_Float16*)take((size_t)NNODES * 64 * 2);
    _Float16* h16b  = (_Float16*)take((size_t)NNODES * 64 * 2);
    float*    m_acc = (float*)take((size_t)NNODES * 64 * 4);
    float*    pool  = (float*)take((size_t)NGRAPHS * 64 * 4);
    _Float16* W1f   = (_Float16*)take(3ULL * 2 * 2 * 17 * 512 * 2);  // 34816 halves/layer
    _Float16* W2f   = (_Float16*)take(3ULL * 9 * 4 * 512 * 2);       // 18432 halves/layer
    _Float16* nW1f  = (_Float16*)take(3ULL * 4 * 8 * 512 * 2);       // 16384 halves/layer
    _Float16* nW2f  = (_Float16*)take(3ULL * 4 * 4 * 512 * 2);       // 8192 halves/layer
    _Float16* l1f   = (_Float16*)take((size_t)2 * 4 * 512 * 2);
    _Float16* l2f   = (_Float16*)take((size_t)2 * 4 * 512 * 2);

    // one-time conversions
    k_h_to_f16<<<(NNODES * 64 + 255) / 256, 256, 0, stream>>>(x, h16a, (long)NNODES * 64);
    k_rel_dist<<<(NEDGES + 255) / 256, 256, 0, stream>>>(pos, srcI, dstI, rel);
    k_conv_eW1<<<(3 * 34816 + 255) / 256, 256, 0, stream>>>(eW1, W1f);
    for (int l = 0; l < 3; ++l) {
        k_conv_bfrag<<<(18432 + 255) / 256, 256, 0, stream>>>(
            eW2 + (size_t)l * 258 * 64, W2f + (size_t)l * 18432, 258, 64, 9, 4);
        k_conv_bfrag<<<(16384 + 255) / 256, 256, 0, stream>>>(
            nW1 + (size_t)l * 128 * 128, nW1f + (size_t)l * 16384, 128, 128, 4, 8);
        k_conv_bfrag<<<(8192 + 255) / 256, 256, 0, stream>>>(
            nW2 + (size_t)l * 128 * 64, nW2f + (size_t)l * 8192, 128, 64, 4, 4);
    }
    k_conv_bfrag<<<16, 256, 0, stream>>>(l1W, l1f, 64, 64, 2, 4);
    k_conv_bfrag<<<16, 256, 0, stream>>>(l2W, l2f, 64, 64, 2, 4);

    hipMemsetAsync(pool, 0, (size_t)NGRAPHS * 64 * 4, stream);

    _Float16* hin = h16a;
    _Float16* hout = h16b;
    const int nodeBlocks = (NNODES + 63) / 64;  // 2 waves x 32 nodes per block
    for (int l = 0; l < 3; ++l) {
        hipMemsetAsync(m_acc, 0, (size_t)NNODES * 64 * 4, stream);
        k_edge<<<NEDGES / 64, 64, 0, stream>>>(
            hin, rel, srcI, dstI,
            eW1 + (size_t)l * 129 * 258, eb1 + (size_t)l * 258, eb2 + (size_t)l * 64,
            W1f + (size_t)l * 34816, W2f + (size_t)l * 18432, m_acc);
        k_node<<<nodeBlocks, 64, 0, stream>>>(
            hin, m_acc, nb1 + (size_t)l * 128, nb2 + (size_t)l * 64,
            nW1f + (size_t)l * 16384, nW2f + (size_t)l * 8192, hout);
        _Float16* t = hin; hin = hout; hout = t;
    }

    k_head_node<<<nodeBlocks, 64, 0, stream>>>(hin, batch, l1b, l2b, l1f, l2f, pool);
    k_head_graph<<<1, 128, 0, stream>>>(pool, l3W, l3b, l4W, l4b, out);
}
